// MoEGatedMLP_57698590655109
// MI455X (gfx1250) — compile-verified, matrix operand order checked by reading, not run on hardware
//
#include <hip/hip_runtime.h>
#include <hip/hip_bf16.h>

// ---------------------------------------------------------------------------
// MoE gated MLP for MI455X (gfx1250).
// bf16 WMMA GEMMs (weights converted once; ~138MB bf16 fits the 192MB L2),
// top-2 gathered expert execution, and CDNA5 async global->LDS staging
// (GLOBAL_LOAD_ASYNC_TO_LDS_B128 + ASYNCcnt) with LDS double buffering.
// ---------------------------------------------------------------------------

typedef __bf16 bf16_t;
typedef bf16_t v16bf __attribute__((ext_vector_type(16)));
typedef bf16_t v8bf  __attribute__((ext_vector_type(8)));
typedef float  v8f   __attribute__((ext_vector_type(8)));

constexpr int NTOK = 2 * 2048;   // B*S
constexpr int HDIM = 1024;
constexpr int IDIM = 2816;
constexpr int NEXP = 7;

#define BM 128
#define BN 128
#define BK 32
#define LDT (BK + 8)   // 40 bf16 = 80B row stride: conflict-free 16B ds_load_b128

// --------------------------------- helpers ---------------------------------

__device__ inline void atomicAddF32(float* p, float v) {
    __hip_atomic_fetch_add(p, v, __ATOMIC_RELAXED, __HIP_MEMORY_SCOPE_AGENT);
}

// 32-bit LDS offset of a __shared__ object (generic -> addrspace(3) -> int)
__device__ __forceinline__ uint32_t lds_addr(const void* p) {
    return (uint32_t)(uintptr_t)(const __attribute__((address_space(3))) void*)p;
}

// CDNA5 async bulk copy: 16B global -> LDS, tracked by ASYNCcnt (no VGPR data)
__device__ __forceinline__ void async_ld_b128(uint32_t lds_off, const void* gptr) {
    asm volatile("global_load_async_to_lds_b128 %0, %1, off"
                 :: "v"(lds_off), "v"((uint64_t)(uintptr_t)gptr)
                 : "memory");
}
#define WAIT_ASYNC(n) asm volatile("s_wait_asynccnt " #n ::: "memory")

// fp32 -> bf16 bulk convert (element counts are multiples of 4 here)
__global__ __launch_bounds__(256) void f32_to_bf16_kernel(
    const float* __restrict__ in, bf16_t* __restrict__ out, long n)
{
    long i = ((long)blockIdx.x * 256 + threadIdx.x) * 4;
    if (i >= n) return;
    float4 v = *(const float4*)(in + i);
    out[i + 0] = (bf16_t)v.x;
    out[i + 1] = (bf16_t)v.y;
    out[i + 2] = (bf16_t)v.z;
    out[i + 3] = (bf16_t)v.w;
}

__global__ void zero_counts_kernel(int* counts) {
    if (threadIdx.x < NEXP) counts[threadIdx.x] = 0;
}

// ------------------------------ router -------------------------------------
// One wave32 per token: 7 dot products over H, shuffle-reduce, top-2 softmax.
__global__ __launch_bounds__(256) void router_kernel(
    const float* __restrict__ x, const float* __restrict__ Wr,
    float* __restrict__ logits, float* __restrict__ comb,
    int* __restrict__ counts, int* __restrict__ toklist)
{
    const int tid = threadIdx.x;
    const int lane = tid & 31, wid = tid >> 5;
    const int n = blockIdx.x * 8 + wid;
    if (n >= NTOK) return;

    const float* xr = x + (size_t)n * HDIM;
    float acc[NEXP];
#pragma unroll
    for (int e = 0; e < NEXP; ++e) acc[e] = 0.f;

    for (int h = lane; h < HDIM; h += 32) {
        float xv = xr[h];
#pragma unroll
        for (int e = 0; e < NEXP; ++e) acc[e] += xv * Wr[e * HDIM + h];
    }
#pragma unroll
    for (int off = 16; off > 0; off >>= 1)
#pragma unroll
        for (int e = 0; e < NEXP; ++e) acc[e] += __shfl_xor(acc[e], off, 32);

    // top-2 (ties -> lower index, like jax top_k)
    int s0 = 0;
#pragma unroll
    for (int e = 1; e < NEXP; ++e) if (acc[e] > acc[s0]) s0 = e;
    int s1 = (s0 == 0) ? 1 : 0;
#pragma unroll
    for (int e = 0; e < NEXP; ++e) if (e != s0 && acc[e] > acc[s1]) s1 = e;

    float e1 = __expf(acc[s1] - acc[s0]);
    float w0 = 1.f / (1.f + e1);
    float w1 = e1 / (1.f + e1);

    if (lane < NEXP) {
        logits[(size_t)n * NEXP + lane] = acc[lane];
        comb[(size_t)n * NEXP + lane] = (lane == s0) ? w0 : (lane == s1) ? w1 : 0.f;
    }
    if (lane == 0) {
        int p0 = atomicAdd(&counts[s0], 1); toklist[s0 * NTOK + p0] = n;
        int p1 = atomicAdd(&counts[s1], 1); toklist[s1 * NTOK + p1] = n;
    }
}

// ------------------------------ WMMA GEMM ----------------------------------
// C[M,N] = A[M,K](bf16, optionally row-gathered) * B[N,K](bf16)^T
// Double-buffered LDS fed by async global->LDS copies (ASYNCcnt).
// MODE 0: store fp32 C      (shared down-projection -> d_out)
// MODE 1: store bf16 C      (gate / up buffers)
// MODE 2: atomicAdd comb[tok,e] * C into out[tok, :]   (expert down-proj)
template<int MODE>
__global__ __launch_bounds__(256) void gemm_bf16_nt(
    const bf16_t* __restrict__ A, const bf16_t* __restrict__ B,
    void* __restrict__ Cout, int Mconst, const int* __restrict__ Mptr,
    int Nout, int K, const int* __restrict__ gather,
    const float* __restrict__ comb, int expert)
{
    const int Mval = Mptr ? *Mptr : Mconst;
    const int m0 = blockIdx.y * BM;
    if (m0 >= Mval) return;                 // uniform early-exit (expert grids fixed-size)
    const int n0 = blockIdx.x * BN;

    __shared__ __align__(16) bf16_t As[2][BM][LDT];   // 2 x 10240 B
    __shared__ __align__(16) bf16_t Bs[2][BN][LDT];   // 2 x 10240 B

    const int tid = threadIdx.x;
    const int lane = tid & 31, wave = tid >> 5;
    const int wm = wave >> 1;               // 0..3 : 32-row slab
    const int wn = wave & 1;                // 0..1 : 64-col slab
    const int half = lane >> 4, l15 = lane & 15;

    // ---- per-thread staging slots: 2 x 16B for A, 2 x 16B for B per tile ----
    const bf16_t* aSrc[2];
    const bf16_t* bSrc[2];
    uint32_t aDst[2], bDst[2];
#pragma unroll
    for (int p = 0; p < 2; ++p) {
        int idx = (p * 256 + tid) * 8;      // 4096 bf16 per tile per matrix
        int row = idx >> 5;                 // / BK
        int col = idx & (BK - 1);
        int gr = m0 + row;
        if (gr >= Mval) gr = Mval - 1;      // clamp keeps gather/load in-bounds
        int arow = gather ? gather[gr] : gr;
        aSrc[p] = A + (size_t)arow * K + col;
        bSrc[p] = B + (size_t)(n0 + row) * K + col;
        aDst[p] = lds_addr(&As[0][row][col]);
        bDst[p] = lds_addr(&Bs[0][row][col]);
    }
    const uint32_t BUFB = (uint32_t)(BM * LDT * sizeof(bf16_t));  // 10240

    auto issue_tile = [&](int buf, int kb) {
#pragma unroll
        for (int p = 0; p < 2; ++p)
            async_ld_b128(aDst[p] + buf * BUFB, aSrc[p] + kb);
#pragma unroll
        for (int p = 0; p < 2; ++p)
            async_ld_b128(bDst[p] + buf * BUFB, bSrc[p] + kb);
    };

    v8f acc[2][4];
#pragma unroll
    for (int i = 0; i < 2; ++i)
#pragma unroll
        for (int j = 0; j < 4; ++j)
#pragma unroll
            for (int v = 0; v < 8; ++v) acc[i][j][v] = 0.f;

    const int KT = K / BK;
    issue_tile(0, 0);

    for (int t = 0; t < KT; ++t) {
        const int cur = t & 1;
        if (t + 1 < KT) {
            issue_tile(cur ^ 1, (t + 1) * BK);  // prefetch next tile into other buffer
            WAIT_ASYNC(4);                       // this wave's current-tile ops done
        } else {
            WAIT_ASYNC(0);
        }
        __syncthreads();                         // all waves' tile data in LDS

        // fragment loads: 16-bit A/B layout, lane<16 -> K{0..7,16..23}, lane>=16 -> +8
        union Frag { v16bf v; v8bf h[2]; } af[2], bfr[4];
#pragma unroll
        for (int mt = 0; mt < 2; ++mt) {
            int r = wm * 32 + mt * 16 + l15;
            af[mt].h[0] = *(const v8bf*)&As[cur][r][half * 8];
            af[mt].h[1] = *(const v8bf*)&As[cur][r][16 + half * 8];
        }
#pragma unroll
        for (int nt = 0; nt < 4; ++nt) {
            int r = wn * 64 + nt * 16 + l15;
            bfr[nt].h[0] = *(const v8bf*)&Bs[cur][r][half * 8];
            bfr[nt].h[1] = *(const v8bf*)&Bs[cur][r][16 + half * 8];
        }
#pragma unroll
        for (int mt = 0; mt < 2; ++mt)
#pragma unroll
            for (int nt = 0; nt < 4; ++nt)
                acc[mt][nt] = __builtin_amdgcn_wmma_f32_16x16x32_bf16(
                    false, af[mt].v, false, bfr[nt].v,
                    (short)0, acc[mt][nt], false, false);

        if (t + 1 < KT) __syncthreads();         // buffer 'cur' free for t+2 prefetch
    }

    // epilogue: C layout -> lane<16 holds rows m..m+7 of col l15; lane>=16 rows +8
#pragma unroll
    for (int mt = 0; mt < 2; ++mt) {
#pragma unroll
        for (int nt = 0; nt < 4; ++nt) {
            int col = n0 + wn * 64 + nt * 16 + l15;
#pragma unroll
            for (int v = 0; v < 8; ++v) {
                int m = m0 + wm * 32 + mt * 16 + half * 8 + v;
                if (m < Mval) {
                    float val = acc[mt][nt][v];
                    if (MODE == 0) {
                        ((float*)Cout)[(size_t)m * Nout + col] = val;
                    } else if (MODE == 1) {
                        ((bf16_t*)Cout)[(size_t)m * Nout + col] = (bf16_t)val;
                    } else {
                        int tok = gather[m];
                        float w = comb[(size_t)tok * NEXP + expert];
                        atomicAddF32(&((float*)Cout)[(size_t)tok * Nout + col], w * val);
                    }
                }
            }
        }
    }
}

// ------------------------------ SwiGLU -------------------------------------
__global__ __launch_bounds__(256) void swiglu_kernel(
    const bf16_t* __restrict__ g, const bf16_t* __restrict__ u,
    bf16_t* __restrict__ act, int Mconst, const int* __restrict__ Mptr)
{
    int Mval = Mptr ? *Mptr : Mconst;
    long idx = ((long)blockIdx.x * 256 + threadIdx.x) * 4;
    if (idx >= (long)Mval * IDIM) return;
#pragma unroll
    for (int t = 0; t < 4; ++t) {
        float gf = (float)g[idx + t];
        float uf = (float)u[idx + t];
        float s = gf / (1.f + __expf(-gf));     // silu(g) * u
        act[idx + t] = (bf16_t)(s * uf);
    }
}

// ------------------------------ launch -------------------------------------

extern "C" void kernel_launch(void* const* d_in, const int* in_sizes, int n_in,
                              void* d_out, int out_size, void* d_ws, size_t ws_size,
                              hipStream_t stream)
{
    const float* x   = (const float*)d_in[0];   // [N,H]
    const float* Wgs = (const float*)d_in[1];   // [I,H]
    const float* Wus = (const float*)d_in[2];   // [I,H]
    const float* Wds = (const float*)d_in[3];   // [H,I]
    const float* Wg  = (const float*)d_in[4];   // [E,I,H]
    const float* Wu  = (const float*)d_in[5];   // [E,I,H]
    const float* Wd  = (const float*)d_in[6];   // [E,H,I]
    const float* Wr  = (const float*)d_in[7];   // [E,H]

    float* out    = (float*)d_out;                    // [N,H]
    float* logits = out + (size_t)NTOK * HDIM;        // [N,E]

    // workspace layout (~216 MB)
    char* ws = (char*)d_ws;
    size_t off = 0;
    auto take = [&](size_t bytes) -> char* {
        char* p = ws + off;
        off = (off + bytes + 255) & ~(size_t)255;
        return p;
    };
    bf16_t* xb  = (bf16_t*)take((size_t)NTOK * HDIM * 2);
    bf16_t* wgs = (bf16_t*)take((size_t)IDIM * HDIM * 2);
    bf16_t* wus = (bf16_t*)take((size_t)IDIM * HDIM * 2);
    bf16_t* wds = (bf16_t*)take((size_t)HDIM * IDIM * 2);
    bf16_t* wge = (bf16_t*)take((size_t)NEXP * IDIM * HDIM * 2);
    bf16_t* wue = (bf16_t*)take((size_t)NEXP * IDIM * HDIM * 2);
    bf16_t* wde = (bf16_t*)take((size_t)NEXP * HDIM * IDIM * 2);
    bf16_t* gbuf = (bf16_t*)take((size_t)NTOK * IDIM * 2);
    bf16_t* ubuf = (bf16_t*)take((size_t)NTOK * IDIM * 2);
    bf16_t* act  = (bf16_t*)take((size_t)NTOK * IDIM * 2);
    float*  comb = (float*)take((size_t)NTOK * NEXP * 4);
    int* toklist = (int*)take((size_t)NEXP * NTOK * 4);
    int* counts  = (int*)take(256);

    dim3 blk(256);
    auto cvt = [&](const float* src, bf16_t* dst, long n) {
        unsigned blocks = (unsigned)((n + 1023) / 1024);
        f32_to_bf16_kernel<<<dim3(blocks), blk, 0, stream>>>(src, dst, n);
    };

    zero_counts_kernel<<<dim3(1), dim3(32), 0, stream>>>(counts);

    cvt(x,   xb,  (long)NTOK * HDIM);
    cvt(Wgs, wgs, (long)IDIM * HDIM);
    cvt(Wus, wus, (long)IDIM * HDIM);
    cvt(Wds, wds, (long)HDIM * IDIM);
    cvt(Wg,  wge, (long)NEXP * IDIM * HDIM);
    cvt(Wu,  wue, (long)NEXP * IDIM * HDIM);
    cvt(Wd,  wde, (long)NEXP * HDIM * IDIM);

    router_kernel<<<dim3(NTOK / 8), blk, 0, stream>>>(x, Wr, logits, comb, counts, toklist);

    const dim3 g_gu(IDIM / BN, NTOK / BM);   // (22, 32): C [N, I]
    const dim3 g_dn(HDIM / BN, NTOK / BM);   // ( 8, 32): C [N, H]
    const dim3 g_sw((NTOK * IDIM) / 1024);

    // shared expert
    gemm_bf16_nt<1><<<g_gu, blk, 0, stream>>>(xb, wgs, gbuf, NTOK, nullptr, IDIM, HDIM, nullptr, nullptr, 0);
    gemm_bf16_nt<1><<<g_gu, blk, 0, stream>>>(xb, wus, ubuf, NTOK, nullptr, IDIM, HDIM, nullptr, nullptr, 0);
    swiglu_kernel<<<g_sw, blk, 0, stream>>>(gbuf, ubuf, act, NTOK, nullptr);
    gemm_bf16_nt<0><<<g_dn, blk, 0, stream>>>(act, wds, out, NTOK, nullptr, HDIM, IDIM, nullptr, nullptr, 0);

    // routed experts (gathered top-2 tokens; blocks beyond device count early-exit)
    for (int e = 0; e < NEXP; ++e) {
        const int* cnt = counts + e;
        const int* tl  = toklist + (size_t)e * NTOK;
        const bf16_t* wg_e = wge + (size_t)e * IDIM * HDIM;
        const bf16_t* wu_e = wue + (size_t)e * IDIM * HDIM;
        const bf16_t* wd_e = wde + (size_t)e * HDIM * IDIM;

        gemm_bf16_nt<1><<<g_gu, blk, 0, stream>>>(xb, wg_e, gbuf, NTOK, cnt, IDIM, HDIM, tl, nullptr, 0);
        gemm_bf16_nt<1><<<g_gu, blk, 0, stream>>>(xb, wu_e, ubuf, NTOK, cnt, IDIM, HDIM, tl, nullptr, 0);
        swiglu_kernel<<<g_sw, blk, 0, stream>>>(gbuf, ubuf, act, NTOK, cnt);
        gemm_bf16_nt<2><<<g_dn, blk, 0, stream>>>(act, wd_e, out, NTOK, cnt, HDIM, IDIM, tl, comb, e);
    }
    (void)in_sizes; (void)n_in; (void)out_size; (void)ws_size;
}
